// InvAttentionGATv2_11948599018110
// MI455X (gfx1250) — matrix operand order repaired and checked by reading, not compiled, not run on hardware
//
#include <hip/hip_runtime.h>
#include <math.h>

typedef __attribute__((ext_vector_type(16))) _Float16 v16h;
typedef __attribute__((ext_vector_type(8)))  float    v8f;

#define IN_CH 128
#define H_CH  128

// -------------------- helpers --------------------

__device__ __forceinline__ void atomicMaxFloat(float* addr, float value) {
    // Sign-aware integer trick: correct for mixed-sign floats with -inf init.
    if (value >= 0.0f) {
        atomicMax((int*)addr, __float_as_int(value));
    } else {
        atomicMin((unsigned int*)addr, __float_as_uint(value));
    }
}

// -------------------- init --------------------

__global__ void init_kernel(float* __restrict__ out, float* __restrict__ seg_max,
                            float* __restrict__ denom, int n_out, int n_nodes) {
    int i = blockIdx.x * blockDim.x + threadIdx.x;
    if (i < n_out) out[i] = 0.0f;
    if (i < n_nodes) {
        seg_max[i] = -INFINITY;
        denom[i]   = 0.0f;
    }
}

// -------------------- GEMM: g = q @ W^T via WMMA f16 (f32 accumulate) ------
// Block = 512 threads = 16 waves. Each block computes rows [row0, row0+16).
// Waves 0..7  -> g_l column tiles 0..7 (W_l), waves 8..15 -> g_r (W_r).
// Each wave: one 16x16 f32 tile, K-loop of 4 x v_wmma_f32_16x16x32_f16.

__global__ void gemm_wmma_kernel(const float* __restrict__ q,
                                 const float* __restrict__ W_l,
                                 const float* __restrict__ W_r,
                                 float* __restrict__ g_l,
                                 float* __restrict__ g_r,
                                 int n_nodes) {
    const int lane = threadIdx.x & 31;
    const int wave = threadIdx.x >> 5;          // 0..15
    const int row0 = blockIdx.x * 16;

    const bool  is_r = (wave >= 8);
    const int   col0 = (wave & 7) * 16;
    const float* __restrict__ W = is_r ? W_r : W_l;
    float* __restrict__       g = is_r ? g_r : g_l;

    const int m  = lane & 15;                   // A row within tile
    const int n  = lane & 15;                   // B/C column within tile
    const int hi = lane >> 4;                   // lane-half select

    int rowA = row0 + m;
    if (rowA >= n_nodes) rowA = n_nodes - 1;    // clamp (N divisible by 16 anyway)

    const float* __restrict__ qrow = q + (size_t)rowA * IN_CH;
    const float* __restrict__ wrow = W + (size_t)(col0 + n) * IN_CH;

    const int aoff = hi * 8;                    // A: K offset per lane-half
    const int boff = hi * 8;                    // B: K offset per lane-half

    v8f acc = {};
#pragma unroll
    for (int kb = 0; kb < IN_CH; kb += 32) {
        v16h a, b;
        // A 16x32 f16 layout: elems 0..7 -> K = kb+aoff+i ; elems 8..15 -> K+16
#pragma unroll
        for (int i = 0; i < 8; ++i) {
            a[i]     = (_Float16)qrow[kb + aoff + i];
            a[i + 8] = (_Float16)qrow[kb + 16 + aoff + i];
        }
        // B 32x16 f16 layout: elems 0..7 -> K = kb+boff+i ; elems 8..15 -> K+16, N = col0+n (B[k][n]=W[n][k])
#pragma unroll
        for (int i = 0; i < 8; ++i) {
            b[i]     = (_Float16)wrow[kb + boff + i];
            b[i + 8] = (_Float16)wrow[kb + 16 + boff + i];
        }
        acc = __builtin_amdgcn_wmma_f32_16x16x32_f16(
            /*neg_a=*/false, a, /*neg_b=*/false, b,
            /*c_mod=*/(short)0, acc, /*reuse_a=*/false, /*reuse_b=*/false);
    }

    // C/D layout: VGPR r, lanes 0-15 -> (M=r, N=lane); lanes 16-31 -> (M=8+r)
#pragma unroll
    for (int r = 0; r < 8; ++r) {
        const int row = row0 + r + hi * 8;
        if (row < n_nodes) g[(size_t)row * H_CH + col0 + n] = acc[r];
    }
}

// -------------------- edge logits + segment max (wave per edge) ------------

__global__ void edge_logit_kernel(const int* __restrict__ src,
                                  const int* __restrict__ dst,
                                  const float* __restrict__ g_l,
                                  const float* __restrict__ g_r,
                                  const float* __restrict__ attn_w,
                                  const float* __restrict__ envelope,
                                  float* __restrict__ e_out,
                                  float* __restrict__ seg_max,
                                  int n_edges) {
    const int gid  = blockIdx.x * blockDim.x + threadIdx.x;
    const int edge = gid >> 5;
    const int lane = threadIdx.x & 31;
    if (edge >= n_edges) return;

    const int s = src[edge];
    const int d = dst[edge];

    const float4 gl = ((const float4*)(g_l + (size_t)s * H_CH))[lane];
    const float4 gr = ((const float4*)(g_r + (size_t)d * H_CH))[lane];
    const float4 aw = ((const float4*)attn_w)[lane];

    float p = 0.0f;
    {
        float x;
        x = gl.x + gr.x; p += aw.x * (x / (1.0f + __expf(-x)));
        x = gl.y + gr.y; p += aw.y * (x / (1.0f + __expf(-x)));
        x = gl.z + gr.z; p += aw.z * (x / (1.0f + __expf(-x)));
        x = gl.w + gr.w; p += aw.w * (x / (1.0f + __expf(-x)));
    }
#pragma unroll
    for (int off = 16; off > 0; off >>= 1) p += __shfl_xor(p, off, 32);

    if (lane == 0) {
        const float ev = p + __logf(envelope[edge] + 1e-7f);
        e_out[edge] = ev;
        atomicMaxFloat(seg_max + d, ev);
    }
}

// -------------------- exp + segment denom (thread per edge) ----------------

__global__ void exp_denom_kernel(const int* __restrict__ dst,
                                 const float* __restrict__ seg_max,
                                 float* __restrict__ e_inout,  // e in, ex out
                                 float* __restrict__ denom,
                                 int n_edges) {
    const int edge = blockIdx.x * blockDim.x + threadIdx.x;
    if (edge >= n_edges) return;
    const int d = dst[edge];
    const float ex = __expf(e_inout[edge] - seg_max[d]);
    e_inout[edge] = ex;
    atomicAdd(denom + d, ex);
}

// -------------------- normalize + scatter accumulate (wave per edge) -------

__global__ void scatter_kernel(const int* __restrict__ src,
                               const int* __restrict__ dst,
                               const float* __restrict__ g_l,
                               const float* __restrict__ ex,
                               const float* __restrict__ denom,
                               float* __restrict__ out,
                               int n_edges) {
    const int gid  = blockIdx.x * blockDim.x + threadIdx.x;
    const int edge = gid >> 5;
    const int lane = threadIdx.x & 31;
    if (edge >= n_edges) return;

    const int s = src[edge];
    const int d = dst[edge];
    const float attn = ex[edge] / denom[d];

    const float4 gl = ((const float4*)(g_l + (size_t)s * H_CH))[lane];
    float* __restrict__ orow = out + (size_t)d * H_CH + lane * 4;
    atomicAdd(orow + 0, attn * gl.x);
    atomicAdd(orow + 1, attn * gl.y);
    atomicAdd(orow + 2, attn * gl.z);
    atomicAdd(orow + 3, attn * gl.w);
}

// -------------------- launch --------------------

extern "C" void kernel_launch(void* const* d_in, const int* in_sizes, int n_in,
                              void* d_out, int out_size, void* d_ws, size_t ws_size,
                              hipStream_t stream) {
    // setup_inputs order: q, k, v, envelope, W_l, W_r, attn_w, edge_index
    const float* q        = (const float*)d_in[0];
    const float* envelope = (const float*)d_in[3];
    const float* W_l      = (const float*)d_in[4];
    const float* W_r      = (const float*)d_in[5];
    const float* attn_w   = (const float*)d_in[6];
    const int*   eidx     = (const int*)d_in[7];

    const int n_nodes = in_sizes[0] / IN_CH;      // 50000
    const int n_edges = in_sizes[7] / 2;          // 800000
    const int* src = eidx;                        // edge_index[0]
    const int* dst = eidx + n_edges;              // edge_index[1]

    float* out = (float*)d_out;

    // workspace layout
    float* g_l     = (float*)d_ws;
    float* g_r     = g_l + (size_t)n_nodes * H_CH;
    float* e_buf   = g_r + (size_t)n_nodes * H_CH;   // e, then ex in-place
    float* seg_max = e_buf + (size_t)n_edges;
    float* denom   = seg_max + (size_t)n_nodes;

    const int n_out = n_nodes * H_CH;

    // 1) init out / seg_max / denom
    {
        int threads = 256;
        int blocks  = (n_out + threads - 1) / threads;
        init_kernel<<<blocks, threads, 0, stream>>>(out, seg_max, denom, n_out, n_nodes);
    }
    // 2) WMMA GEMM: g_l, g_r
    {
        int blocks = (n_nodes + 15) / 16;         // 3125
        gemm_wmma_kernel<<<blocks, 512, 0, stream>>>(q, W_l, W_r, g_l, g_r, n_nodes);
    }
    // 3) edge logits + segment max (wave per edge)
    {
        int threads = 256;                        // 8 waves / block
        int blocks  = (n_edges + 7) / 8;
        edge_logit_kernel<<<blocks, threads, 0, stream>>>(src, dst, g_l, g_r, attn_w,
                                                          envelope, e_buf, seg_max, n_edges);
    }
    // 4) exp + denom (thread per edge)
    {
        int threads = 256;
        int blocks  = (n_edges + threads - 1) / threads;
        exp_denom_kernel<<<blocks, threads, 0, stream>>>(dst, seg_max, e_buf, denom, n_edges);
    }
    // 5) normalize + scatter (wave per edge)
    {
        int threads = 256;
        int blocks  = (n_edges + 7) / 8;
        scatter_kernel<<<blocks, threads, 0, stream>>>(src, dst, g_l, e_buf, denom,
                                                       out, n_edges);
    }
}
